// Mamba1DCore_60902636257674
// MI455X (gfx1250) — matrix operand order repaired and checked by reading, not compiled, not run on hardware
//
#include <hip/hip_runtime.h>

// Mamba1DCore on MI455X (gfx1250) -- round 3.
// Reference dead-code analysis: dt branch unused; scan ys == its input, so the
// A/carry recurrence is discarded. Output = (conv1d_SAME(x,K) + conv_b) * D.
// Computed as one GEMM: M=2048, N=2048, K=4*2048=8192,
//   A[l, w*2048+i] = xpad[l+w-1, i]  (Toeplitz view, boundary zeros)
//   B[w*2048+i, o] = conv_kernel[w,i,o]
// Precision: bf16x3 split (hi/lo) on v_wmma_f32_16x16x32_bf16, f32 accum.
// v3: prepass converts x / conv_kernel to bf16 hi/lo planes once into d_ws;
// main GEMM stages A with global_load_async_to_lds_b128 (ASYNCcnt) and stages
// B with the Tensor Data Mover (tensor_load_to_lds, TENSORcnt) -- each block's
// per-chunk B panel is a contiguous 4KB region, and TDM pad_enable inserts the
// 48B LDS row padding (64B data + 48B pad = 112B conflict-tuned pitch).
// Double-buffered LDS overlaps both copy engines with the WMMA stream.
// Falls back to the self-contained round-1 kernel if ws_size is too small.

typedef __attribute__((ext_vector_type(16))) __bf16 v16bf;
typedef __attribute__((ext_vector_type(8)))  __bf16 v8bf;
typedef __attribute__((ext_vector_type(8)))  float  v8f;
typedef __attribute__((ext_vector_type(4)))  float  f32x4;
typedef __attribute__((ext_vector_type(4)))  unsigned v4u;
typedef __attribute__((ext_vector_type(8)))  unsigned v8u;

#define SEQ_L   2048
#define DMODEL  2048
#define DCONV   4
#define BM      128
#define BN      64
#define KC      32
#define NCHUNK  ((DCONV * DMODEL) / KC)   // 256 K-chunks
#define CHPW    (DMODEL / KC)             // 64 chunks per conv tap w
#define PITCH   56                        // LDS row pitch in bf16 (112B)
#define PITCH8  (PITCH / 8)

// LDS buffer layout (bytes), per pipeline buffer
#define AHI_O   0
#define ALO_O   14336                     // 128 rows * 112B
#define BHI_O   28672
#define BLO_O   35840                     // + 64 rows * 112B
#define BUFSZ   43008                     // total per buffer; x2 = 86016B

union U8  { v8bf  v; unsigned short s[8]; };
union V16 { v16bf v; v8bf h[2]; };

__device__ __forceinline__ unsigned short bf16_rne(float f) {
  unsigned int u = __float_as_uint(f);
  u += 0x7FFFu + ((u >> 16) & 1u);
  return (unsigned short)(u >> 16);
}
__device__ __forceinline__ float bf16_up(unsigned short h) {
  return __uint_as_float((unsigned int)h << 16);
}

// ---- CDNA5 async copy: 16B global -> LDS, tracked by ASYNCcnt ----
__device__ __forceinline__ void async_ld16(void* lds, const void* g) {
  auto l3 = (__attribute__((address_space(3))) char*)lds;
  asm volatile("global_load_async_to_lds_b128 %0, %1, off"
               :: "v"(l3), "v"(g) : "memory");
}
template <int N>
__device__ __forceinline__ void wait_async() {
  asm volatile("s_wait_asynccnt %0" :: "i"(N) : "memory");
}

// ---- CDNA5 Tensor Data Mover: descriptor-driven global -> LDS DMA ----
__device__ __forceinline__ unsigned lds_off(void* p) {
  return (unsigned)(__UINTPTR_TYPE__)(__attribute__((address_space(3))) char*)p;
}
// D# group 0 (ISA 8.3): count=1 | lds_addr | global_addr | type=2
__device__ __forceinline__ v4u tdm_g0(unsigned lds_addr, const void* gaddr) {
  unsigned long long ga = (unsigned long long)(__UINTPTR_TYPE__)gaddr;
  v4u g;
  g.x = 1u;                                   // count=1, user descriptor
  g.y = lds_addr;                             // bits 63:32  lds_addr
  g.z = (unsigned)ga;                         // bits 95:64  global_addr lo
  g.w = ((unsigned)(ga >> 32) & 0x01FFFFFFu)  // bits 120:96 global_addr hi
        | (2u << 30);                         // bits 127:126 type=2 (image)
  return g;
}
__device__ __forceinline__ void tdm_load(v4u g0, v8u g1) {
  asm volatile("tensor_load_to_lds %0, %1" :: "s"(g0), "s"(g1) : "memory");
}

// =====================  prepass: x -> bf16 hi/lo planes  =====================
__global__ __launch_bounds__(256)
void prep_x(const float* __restrict__ x,
            unsigned short* __restrict__ xhi,
            unsigned short* __restrict__ xlo,
            unsigned short* __restrict__ zp) {
  const int t = threadIdx.x;
  if (blockIdx.x == 0 && t < 16) {   // zero page for OOB boundary rows
    U8 z;
    #pragma unroll
    for (int j = 0; j < 8; ++j) z.s[j] = 0;
    *(v8bf*)(zp + t * 8) = z.v;
  }
  const size_t base = ((size_t)blockIdx.x * 256 + t) * 8;
  const f32x4* src = reinterpret_cast<const f32x4*>(x + base);
  f32x4 f0 = src[0], f1 = src[1];
  float fv[8] = {f0.x, f0.y, f0.z, f0.w, f1.x, f1.y, f1.z, f1.w};
  U8 hi, lo;
  #pragma unroll
  for (int j = 0; j < 8; ++j) {
    unsigned short hs = bf16_rne(fv[j]);
    hi.s[j] = hs;
    lo.s[j] = bf16_rne(fv[j] - bf16_up(hs));
  }
  *(v8bf*)(xhi + base) = hi.v;
  *(v8bf*)(xlo + base) = lo.v;
}

// ==========  prepass: conv_kernel -> transposed chunk-major bf16  ===========
// Output layout: [kc][n(2048)][kk(32)] bf16 -> per (block,chunk) B panel is a
// contiguous 64-row x 64B = 4KB region (TDM-friendly).
__global__ __launch_bounds__(256)
void prep_b(const float* __restrict__ convk,
            unsigned short* __restrict__ bthi,
            unsigned short* __restrict__ btlo) {
  __shared__ float tile[32 * 256];
  const int t = threadIdx.x;
  const int kc = blockIdx.x;
  const int n0 = blockIdx.y * 256;
  #pragma unroll
  for (int i = 0; i < 8; ++i) {            // coalesced read 32x256 f32 tile
    int v = t + i * 256;
    int k = v >> 6, c = (v & 63) * 4;
    const f32x4* s = reinterpret_cast<const f32x4*>(
        convk + (size_t)(kc * KC + k) * DMODEL + n0 + c);
    *(f32x4*)&tile[k * 256 + c] = s[0];
  }
  __syncthreads();
  U8 hi[4], lo[4];
  #pragma unroll
  for (int k = 0; k < 32; ++k) {           // gather column n = t (no conflicts)
    float f = tile[k * 256 + t];
    unsigned short hs = bf16_rne(f);
    hi[k >> 3].s[k & 7] = hs;
    lo[k >> 3].s[k & 7] = bf16_rne(f - bf16_up(hs));
  }
  const size_t off = ((size_t)kc * DMODEL + n0 + t) * 32;
  #pragma unroll
  for (int s = 0; s < 4; ++s) {            // coalesced 64B/plane per thread
    *(v8bf*)(bthi + off + s * 8) = hi[s].v;
    *(v8bf*)(btlo + off + s * 8) = lo[s].v;
  }
}

// ==========  main GEMM: async + TDM staged, double-buffered LDS  ============
__global__ __launch_bounds__(256)
void Mamba1DCore_conv_wmma_async(const char* __restrict__ xhi,
                                 const char* __restrict__ xlo,
                                 const char* __restrict__ bthi,
                                 const char* __restrict__ btlo,
                                 const char* __restrict__ zpad,
                                 const float* __restrict__ convb,
                                 const float* __restrict__ dparam,
                                 float* __restrict__ out) {
  extern __shared__ char smem[];           // 2 * BUFSZ bytes
  const int t     = threadIdx.x;
  const int lane  = t & 31;
  const int wid   = t >> 5;                // 8 waves (wave32)
  const int wm    = wid >> 1, wn = wid & 1;
  const int lhalf = lane >> 4, l15 = lane & 15;
  const int blockN = blockIdx.x * BN;
  const int blockM = blockIdx.y * BM;

  // D# group 1 (ISA 8.4), constant across chunks:
  //   data_size=4B (code 2); tile 16 elems x 64 rows; tensor_dim0=16,
  //   tensor_dim1=64, dim0_stride=16 (rows contiguous);
  //   pad_enable, pad_interval=16 DWORDs (code 3), pad_amount=12 DWORDs
  //   (code 11): 64B data + 48B pad = 112B LDS pitch.
  v8u g1;
  g1[0] = (2u << 16) | (1u << 20) | (3u << 22) | (11u << 25);
  g1[1] = 16u << 16;   // tensor_dim0[15:0] in bits 63:48
  g1[2] = 64u << 16;   // tensor_dim1[15:0] in bits 95:80
  g1[3] = 16u << 16;   // tile_dim0 in bits 127:112
  g1[4] = 64u;         // tile_dim1
  g1[5] = 16u;         // tensor_dim0_stride[31:0]
  g1[6] = 0u;
  g1[7] = 0u;

  auto stage = [&](int kc, int buf) {
    const int w  = kc / CHPW;
    const int i0 = (kc % CHPW) * KC;
    char* base = smem + (size_t)buf * BUFSZ;
    // ---- A: 128x32 hi+lo via per-lane async b128 (Toeplitz shift + zeros)
    #pragma unroll
    for (int it = 0; it < 2; ++it) {
      int slot = t + it * 256;
      int row  = slot >> 2, seg = slot & 3;
      int r    = blockM + row + w - 1;     // SAME pad lo=1
      const char *gh, *gl;
      if ((unsigned)r < (unsigned)SEQ_L) {
        size_t bo = ((size_t)r * DMODEL + i0 + seg * 8) * 2;
        gh = xhi + bo; gl = xlo + bo;
      } else {
        gh = zpad; gl = zpad;              // boundary rows read zeros
      }
      char* l = base + row * 112 + seg * 16;
      async_ld16(l + AHI_O, gh);
      async_ld16(l + ALO_O, gl);
    }
    // ---- B: contiguous 4KB panels via TDM, one issuing wave only
    // (TDM ignores EXEC; gate by wave so exactly one DMA per descriptor)
    if (wid == 0) {
      size_t bo = ((size_t)kc * DMODEL + blockN) * 64;
      tdm_load(tdm_g0(lds_off(base + BHI_O), bthi + bo), g1);
      tdm_load(tdm_g0(lds_off(base + BLO_O), btlo + bo), g1);
    }
  };

  const v8f z8 = {0.f, 0.f, 0.f, 0.f, 0.f, 0.f, 0.f, 0.f};
  v8f acc[2][2];
  acc[0][0] = z8; acc[0][1] = z8; acc[1][0] = z8; acc[1][1] = z8;

  stage(0, 0);                             // prologue copies

  for (int kc = 0; kc < NCHUNK; ++kc) {
    const int cur = kc & 1;
    if (kc + 1 < NCHUNK) {
      stage(kc + 1, cur ^ 1);              // copies overlap this chunk's WMMAs
      wait_async<4>();                     // oldest 4 A-copies (cur buf) done
      __builtin_amdgcn_s_wait_tensorcnt((short)2);  // cur buf's 2 TDM ops done
    } else {
      wait_async<0>();
      __builtin_amdgcn_s_wait_tensorcnt((short)0);
    }
    __syncthreads();                       // all waves' copies visible

    const char* bufp = smem + (size_t)cur * BUFSZ;
    V16 ah[2], al[2], bfh[2], bfl[2];
    #pragma unroll
    for (int mt = 0; mt < 2; ++mt) {       // A frags (ISA 7.12.2 layout)
      int rowm = wm * 32 + mt * 16 + l15;
      const v8bf* pa = (const v8bf*)__builtin_assume_aligned(
          bufp + AHI_O + rowm * 112, 16);
      const v8bf* pl = (const v8bf*)__builtin_assume_aligned(
          bufp + ALO_O + rowm * 112, 16);
      ah[mt].h[0] = pa[lhalf];     ah[mt].h[1] = pa[lhalf + 2];
      al[mt].h[0] = pl[lhalf];     al[mt].h[1] = pl[lhalf + 2];
    }
    #pragma unroll
    for (int nt = 0; nt < 2; ++nt) {       // B frags
      int rown = wn * 32 + nt * 16 + l15;
      const v8bf* ph = (const v8bf*)__builtin_assume_aligned(
          bufp + BHI_O + rown * 112, 16);
      const v8bf* pl = (const v8bf*)__builtin_assume_aligned(
          bufp + BLO_O + rown * 112, 16);
      bfh[nt].h[0] = ph[lhalf];  bfh[nt].h[1] = ph[lhalf + 2];
      bfl[nt].h[0] = pl[lhalf];  bfl[nt].h[1] = pl[lhalf + 2];
    }

    #pragma unroll
    for (int mt = 0; mt < 2; ++mt)
      #pragma unroll
      for (int nt = 0; nt < 2; ++nt) {     // bf16x3: hi*hi + hi*lo + lo*hi
        acc[mt][nt] = __builtin_amdgcn_wmma_f32_16x16x32_bf16(
            false, ah[mt].v, false, bfh[nt].v, (short)0, acc[mt][nt], false, false);
        acc[mt][nt] = __builtin_amdgcn_wmma_f32_16x16x32_bf16(
            false, ah[mt].v, false, bfl[nt].v, (short)0, acc[mt][nt], false, false);
        acc[mt][nt] = __builtin_amdgcn_wmma_f32_16x16x32_bf16(
            false, al[mt].v, false, bfh[nt].v, (short)0, acc[mt][nt], false, false);
      }
    __syncthreads();                       // buffer free for next stage()
  }

  #pragma unroll
  for (int mt = 0; mt < 2; ++mt)
    #pragma unroll
    for (int nt = 0; nt < 2; ++nt) {
      int n  = blockN + wn * 32 + nt * 16 + l15;
      int m0 = blockM + wm * 32 + mt * 16 + lhalf * 8;
      float cb = convb[n], dp = dparam[n];
      #pragma unroll
      for (int r = 0; r < 8; ++r)
        out[(size_t)(m0 + r) * DMODEL + n] = (acc[mt][nt][r] + cb) * dp;
    }
}

// =================  fallback (round-1 self-contained kernel)  ================
__global__ __launch_bounds__(256)
void Mamba1DCore_conv_wmma_fallback(const float* __restrict__ x,
                                    const float* __restrict__ convk,
                                    const float* __restrict__ convb,
                                    const float* __restrict__ dparam,
                                    float* __restrict__ out) {
  __shared__ v8bf sAhi[BM * PITCH8];
  __shared__ v8bf sAlo[BM * PITCH8];
  __shared__ v8bf sBhi[BN * PITCH8];
  __shared__ v8bf sBlo[BN * PITCH8];
  const int t = threadIdx.x, lane = t & 31, wid = t >> 5;
  const int wm = wid >> 1, wn = wid & 1, lhalf = lane >> 4, l15 = lane & 15;
  const int blockN = blockIdx.x * BN, blockM = blockIdx.y * BM;
  const v8f z8 = {0.f, 0.f, 0.f, 0.f, 0.f, 0.f, 0.f, 0.f};
  v8f acc[2][2];
  acc[0][0] = z8; acc[0][1] = z8; acc[1][0] = z8; acc[1][1] = z8;
  for (int kc = 0; kc < NCHUNK; ++kc) {
    const int w = kc / CHPW, i0 = (kc % CHPW) * KC;
    __syncthreads();
    #pragma unroll
    for (int it = 0; it < 2; ++it) {
      int slot = t + it * 256, row = slot >> 2, seg = slot & 3;
      int r = blockM + row + w - 1;
      U8 hi, lo;
      if ((unsigned)r < (unsigned)SEQ_L) {
        const f32x4* src = reinterpret_cast<const f32x4*>(
            x + (size_t)r * DMODEL + i0 + seg * 8);
        f32x4 f0 = src[0], f1 = src[1];
        float fv[8] = {f0.x, f0.y, f0.z, f0.w, f1.x, f1.y, f1.z, f1.w};
        #pragma unroll
        for (int j = 0; j < 8; ++j) {
          unsigned short hs = bf16_rne(fv[j]);
          hi.s[j] = hs;
          lo.s[j] = bf16_rne(fv[j] - bf16_up(hs));
        }
      } else {
        #pragma unroll
        for (int j = 0; j < 8; ++j) { hi.s[j] = 0; lo.s[j] = 0; }
      }
      sAhi[row * PITCH8 + seg] = hi.v;
      sAlo[row * PITCH8 + seg] = lo.v;
    }
    {
      int flat = t * 8, kk = flat >> 6, n0 = flat & 63;
      const f32x4* src = reinterpret_cast<const f32x4*>(
          convk + (size_t)(w * DMODEL + i0 + kk) * DMODEL + blockN + n0);
      f32x4 f0 = src[0], f1 = src[1];
      float fv[8] = {f0.x, f0.y, f0.z, f0.w, f1.x, f1.y, f1.z, f1.w};
      unsigned short* bsh = (unsigned short*)sBhi;
      unsigned short* bsl = (unsigned short*)sBlo;
      #pragma unroll
      for (int j = 0; j < 8; ++j) {
        unsigned short hs = bf16_rne(fv[j]);
        bsh[(n0 + j) * PITCH + kk] = hs;
        bsl[(n0 + j) * PITCH + kk] = bf16_rne(fv[j] - bf16_up(hs));
      }
    }
    __syncthreads();
    V16 ah[2], al[2], bfh[2], bfl[2];
    #pragma unroll
    for (int mt = 0; mt < 2; ++mt) {
      int base = (wm * 32 + mt * 16 + l15) * PITCH8 + lhalf;
      ah[mt].h[0] = sAhi[base];  ah[mt].h[1] = sAhi[base + 2];
      al[mt].h[0] = sAlo[base];  al[mt].h[1] = sAlo[base + 2];
    }
    #pragma unroll
    for (int nt = 0; nt < 2; ++nt) {
      int base = (wn * 32 + nt * 16 + l15) * PITCH8 + lhalf;
      bfh[nt].h[0] = sBhi[base]; bfh[nt].h[1] = sBhi[base + 2];
      bfl[nt].h[0] = sBlo[base]; bfl[nt].h[1] = sBlo[base + 2];
    }
    #pragma unroll
    for (int mt = 0; mt < 2; ++mt)
      #pragma unroll
      for (int nt = 0; nt < 2; ++nt) {
        acc[mt][nt] = __builtin_amdgcn_wmma_f32_16x16x32_bf16(
            false, ah[mt].v, false, bfh[nt].v, (short)0, acc[mt][nt], false, false);
        acc[mt][nt] = __builtin_amdgcn_wmma_f32_16x16x32_bf16(
            false, ah[mt].v, false, bfl[nt].v, (short)0, acc[mt][nt], false, false);
        acc[mt][nt] = __builtin_amdgcn_wmma_f32_16x16x32_bf16(
            false, al[mt].v, false, bfh[nt].v, (short)0, acc[mt][nt], false, false);
      }
  }
  #pragma unroll
  for (int mt = 0; mt < 2; ++mt)
    #pragma unroll
    for (int nt = 0; nt < 2; ++nt) {
      int n  = blockN + wn * 32 + nt * 16 + l15;
      int m0 = blockM + wm * 32 + mt * 16 + lhalf * 8;
      float cb = convb[n], dp = dparam[n];
      #pragma unroll
      for (int r = 0; r < 8; ++r)
        out[(size_t)(m0 + r) * DMODEL + n] = (acc[mt][nt][r] + cb) * dp;
    }
}

extern "C" void kernel_launch(void* const* d_in, const int* in_sizes, int n_in,
                              void* d_out, int out_size, void* d_ws, size_t ws_size,
                              hipStream_t stream) {
  // inputs: x, dt_W, dt_b, conv_kernel, conv_b, A_param, D_param
  // (dt_W/dt_b/A_param are dead in the reference graph and never read)
  const float* x      = (const float*)d_in[0];
  const float* convk  = (const float*)d_in[3];
  const float* convb  = (const float*)d_in[4];
  const float* dparam = (const float*)d_in[6];
  float* out = (float*)d_out;
  (void)in_sizes; (void)n_in; (void)out_size;

  const size_t XPLANE = (size_t)SEQ_L * DMODEL * 2;           // 8 MB
  const size_t BPLANE = (size_t)DCONV * DMODEL * DMODEL * 2;  // 32 MB
  const size_t XHI = 0, XLO = XPLANE, BTHI = 2 * XPLANE,
               BTLO = 2 * XPLANE + BPLANE, ZP = 2 * XPLANE + 2 * BPLANE;
  const size_t need = ZP + 256;

  dim3 grid(DMODEL / BN, SEQ_L / BM);  // (32, 16)

  if (ws_size >= need && d_ws != nullptr) {
    char* ws = (char*)d_ws;
    prep_x<<<dim3((SEQ_L * DMODEL) / (256 * 8)), dim3(256), 0, stream>>>(
        x, (unsigned short*)(ws + XHI), (unsigned short*)(ws + XLO),
        (unsigned short*)(ws + ZP));
    prep_b<<<dim3(NCHUNK, DMODEL / 256), dim3(256), 0, stream>>>(
        convk, (unsigned short*)(ws + BTHI), (unsigned short*)(ws + BTLO));
    Mamba1DCore_conv_wmma_async<<<grid, dim3(256), 2 * BUFSZ, stream>>>(
        ws + XHI, ws + XLO, ws + BTHI, ws + BTLO, ws + ZP, convb, dparam, out);
  } else {
    Mamba1DCore_conv_wmma_fallback<<<grid, dim3(256), 0, stream>>>(
        x, convk, convb, dparam, out);
  }
}
